// VisionAttention_78331613545188
// MI455X (gfx1250) — matrix operand order repaired and checked by reading, not compiled, not run on hardware
//
#include <hip/hip_runtime.h>
#include <stdint.h>

typedef __attribute__((ext_vector_type(16))) __bf16        v16bf;
typedef __attribute__((ext_vector_type(8)))  float         v8f;
typedef __attribute__((ext_vector_type(4)))  unsigned int  u32x4;
typedef __attribute__((ext_vector_type(8)))  int           i32x8;
typedef __attribute__((ext_vector_type(4)))  int           i32x4;

#define S_LEN 4096
#define EMB   1280
#define NH    16
#define HD    80
#define HDP   96   // head dim padded to multiple of 32 for bf16 WMMA K

#if defined(__HIP_DEVICE_COMPILE__) && __has_builtin(__builtin_amdgcn_tensor_load_to_lds) && __has_builtin(__builtin_amdgcn_s_wait_tensorcnt)
#define USE_TDM 1
#else
#define USE_TDM 0
#endif

static __device__ __forceinline__ unsigned short f2bf(float f) {
    unsigned int u = __float_as_uint(f);
    unsigned int r = u + 0x7FFFu + ((u >> 16) & 1u);   // round-to-nearest-even
    return (unsigned short)(r >> 16);
}
static __device__ __forceinline__ float bf2f(unsigned short s) {
    return __uint_as_float(((unsigned int)s) << 16);
}

union FragBF {          // one 16-element bf16 WMMA operand = 8 VGPRs = 2 x b128
    v16bf v;
    u32x4 q[2];
};

#if USE_TDM
// LDS byte offset of a __shared__ object: flat shared-aperture addresses carry the
// wave-relative LDS offset in addr[31:0] (ISA 10.2 aperture mapping).
static __device__ __forceinline__ unsigned lds_off(const void* p) {
    return (unsigned)(size_t)p;
}
// Issue one TDM tile load: elements are 2 bytes. x = tx contiguous elems,
// y = ty rows with stride sy elems, optional z = tz planes with stride sz elems.
// LDS padding (pad_interval/pad_amount codes ic/ac) reproduces the
// bank-decorrelated LDS strides used by the fragment readers.
static __device__ __forceinline__ void tdm_load_tile(
    const void* gptr, unsigned ldsb,
    unsigned tx, unsigned ty, unsigned tz,
    unsigned sy, unsigned sz,
    unsigned ic, unsigned ac)
{
    unsigned long long ga = (unsigned long long)(size_t)gptr;
    u32x4 g0 = { 1u,                                    // count=1 (valid, user mode)
                 ldsb,                                  // lds_addr
                 (unsigned)ga,                          // global_addr[31:0]
                 (unsigned)(ga >> 32) | 0x80000000u };  // addr[56:32] | type=2<<30
    i32x8 g1;
    g1[0] = (int)((1u << 16) | (1u << 20) | (ic << 22) | (ac << 25)); // data_size=2B, pad_enable
    g1[1] = (int)(tx << 16);                   // tensor_dim0[15:0] @ [63:48]
    g1[2] = (int)(ty << 16);                   // tensor_dim0 hi=0; tensor_dim1[15:0] @ [95:80]
    g1[3] = (int)(tx << 16);                   // tensor_dim1 hi=0; tile_dim0 @ [127:112]
    g1[4] = (int)(ty | (tz << 16));            // tile_dim1, tile_dim2
    g1[5] = (int)sy;                           // tensor_dim0_stride low32
    g1[6] = (int)((sz & 0xFFFFu) << 16);       // dim0_stride hi=0; dim1_stride low16
    g1[7] = (int)(sz >> 16);                   // dim1_stride high32
    i32x4 g2 = { (int)tz, 0, 0, 0 };           // tensor_dim2 (0 if unused)
    i32x4 g3 = { 0, 0, 0, 0 };
    i32x8 g4 = { 0, 0, 0, 0, 0, 0, 0, 0 };     // unused group (6-arg builtin form)
    __builtin_amdgcn_tensor_load_to_lds(g0, g1, g2, g3, g4, 0);
}
#endif

// ---------------------------------------------------------------- converts
__global__ void k_f32_to_bf16(const float* __restrict__ in,
                              unsigned short* __restrict__ out, int n) {
    int i = blockIdx.x * blockDim.x + threadIdx.x;
    if (i < n) out[i] = f2bf(in[i]);
}

// ---------------------------------------------------------------- GEMM
// C[M,N] = A[M,K] * B[N,K]^T (+ bias) (+ residual).  K % 32 == 0, N % 128 == 0, M % 128 == 0.
// Block tile 128x128, 8 waves of 64x32; A/B tiles staged by double-buffered TDM DMA.
template<int RESID>
__global__ __launch_bounds__(256)
void k_gemm_bf16(const unsigned short* __restrict__ A,
                 const unsigned short* __restrict__ B,
                 const float* __restrict__ bias,
                 const float* __restrict__ resid,
                 unsigned short* __restrict__ Cbf,
                 float* __restrict__ Cf,
                 int M, int N, int K)
{
    constexpr int BM = 128, BN = 128, BK = 32;
    constexpr int LDA = BK + 8;                 // 40 elems = 80 B: 20-bank stride (TDM pad 4DW/16DW)
    __shared__ alignas(16) unsigned short As[2][BM * LDA];
    __shared__ alignas(16) unsigned short Bs[2][BN * LDA];

    const int tid   = threadIdx.x;
    const int lane  = tid & 31;
    const int wave  = tid >> 5;                 // 8 waves: 2 (M) x 4 (N)
    const int wm    = wave >> 2;                // 0..1 -> 64 rows each
    const int wn    = wave & 3;                 // 0..3 -> 32 cols each
    const int laneM = lane & 15;
    const int laneH = lane >> 4;
    const int m0    = blockIdx.y * BM;
    const int n0    = blockIdx.x * BN;

    v8f acc[4][2] = {};

#if USE_TDM
    if (wave == 0) {    // prologue: DMA k-slab 0 into buffer 0
        tdm_load_tile(A + (size_t)m0 * K, lds_off(&As[0][0]), 32, BM, 0, (unsigned)K, 0, 3, 3);
        tdm_load_tile(B + (size_t)n0 * K, lds_off(&Bs[0][0]), 32, BN, 0, (unsigned)K, 0, 3, 3);
        __builtin_amdgcn_s_wait_tensorcnt(0);
    }
    __syncthreads();
#endif

    for (int k0 = 0; k0 < K; k0 += BK) {
        const int buf = (k0 >> 5) & 1;
#if USE_TDM
        if ((k0 + BK < K) && wave == 0) {       // DMA next slab while WMMAs run
            tdm_load_tile(A + (size_t)m0 * K + k0 + BK, lds_off(&As[buf ^ 1][0]),
                          32, BM, 0, (unsigned)K, 0, 3, 3);
            tdm_load_tile(B + (size_t)n0 * K + k0 + BK, lds_off(&Bs[buf ^ 1][0]),
                          32, BN, 0, (unsigned)K, 0, 3, 3);
        }
#else
        __syncthreads();
        {   // fallback: per-thread staging
            const int lrow = tid >> 2;
            const int lcol = (tid & 3) * 8;
            *(u32x4*)(&As[buf][lrow * LDA + lcol]) =
                *(const u32x4*)(A + (size_t)(m0 + lrow) * K + k0 + lcol);
            *(u32x4*)(&As[buf][(lrow + 64) * LDA + lcol]) =
                *(const u32x4*)(A + (size_t)(m0 + lrow + 64) * K + k0 + lcol);
            *(u32x4*)(&Bs[buf][lrow * LDA + lcol]) =
                *(const u32x4*)(B + (size_t)(n0 + lrow) * K + k0 + lcol);
            *(u32x4*)(&Bs[buf][(lrow + 64) * LDA + lcol]) =
                *(const u32x4*)(B + (size_t)(n0 + lrow + 64) * K + k0 + lcol);
        }
        __syncthreads();
#endif

        FragBF a[4], b[2];
#pragma unroll
        for (int i = 0; i < 4; ++i) {           // A frag: lane=M row, K 0..7/16..23 (+8 hi half)
            const unsigned short* p = &As[buf][(wm * 64 + i * 16 + laneM) * LDA + laneH * 8];
            a[i].q[0] = *(const u32x4*)(p);
            a[i].q[1] = *(const u32x4*)(p + 16);
        }
#pragma unroll
        for (int j = 0; j < 2; ++j) {           // B frag: lane=N row, K 0..7/16..23 (+8 hi half)
            const unsigned short* p = &Bs[buf][(wn * 32 + j * 16 + laneM) * LDA + laneH * 8];
            b[j].q[0] = *(const u32x4*)(p);
            b[j].q[1] = *(const u32x4*)(p + 16);
        }
#pragma unroll
        for (int i = 0; i < 4; ++i)
#pragma unroll
            for (int j = 0; j < 2; ++j)
                acc[i][j] = __builtin_amdgcn_wmma_f32_16x16x32_bf16(
                    false, a[i].v, false, b[j].v, (short)0, acc[i][j], false, false);

#if USE_TDM
        if (wave == 0) __builtin_amdgcn_s_wait_tensorcnt(0);
        __syncthreads();
#endif
    }

#pragma unroll
    for (int i = 0; i < 4; ++i) {
#pragma unroll
        for (int j = 0; j < 2; ++j) {
            const int n  = n0 + wn * 32 + j * 16 + laneM;
            const float bn = bias[n];
#pragma unroll
            for (int r = 0; r < 8; ++r) {
                const int m = m0 + wm * 64 + i * 16 + r + laneH * 8;
                float v = acc[i][j][r] + bn;
                if (RESID) Cf[(size_t)m * N + n] = resid[(size_t)m * N + n] + v;
                else       Cbf[(size_t)m * N + n] = f2bf(v);
            }
        }
    }
}

// ---------------------------------------------------------------- RoPE + layout prep
__global__ void k_rope_prep(const unsigned short* __restrict__ qkv,
                            const float* __restrict__ cosb,
                            const float* __restrict__ sinb,
                            unsigned short* __restrict__ Qp,
                            unsigned short* __restrict__ Kp,
                            unsigned short* __restrict__ Vt)
{
    int idx = blockIdx.x * blockDim.x + threadIdx.x;
    if (idx >= S_LEN * EMB) return;
    int s = idx / EMB;
    int r = idx - s * EMB;
    int h = r / HD;
    int d = r - h * HD;

    const size_t base = (size_t)s * (3 * EMB) + h * HD;
    float qv = bf2f(qkv[base + d]);
    float kv = bf2f(qkv[base + EMB + d]);
    unsigned short vv = qkv[base + 2 * EMB + d];

    int drot = (d < HD / 2) ? (d + HD / 2) : (d - HD / 2);
    float qr = bf2f(qkv[base + drot]);
    float kr = bf2f(qkv[base + EMB + drot]);
    if (d < HD / 2) { qr = -qr; kr = -kr; }

    float c  = cosb[s * HD + d];
    float sn = sinb[s * HD + d];
    const float scale = 0.11180339887498949f * 1.4426950408889634f; // log2e/sqrt(80)
    float qo = (qv * c + qr * sn) * scale;
    float ko = (kv * c + kr * sn);

    size_t o = ((size_t)h * S_LEN + s) * HDP + d;
    Qp[o] = f2bf(qo);
    Kp[o] = f2bf(ko);
    if (d < HDP - HD) {                          // zero pad cols 80..95
        Qp[((size_t)h * S_LEN + s) * HDP + HD + d] = 0;
        Kp[((size_t)h * S_LEN + s) * HDP + HD + d] = 0;
    }
    Vt[((size_t)h * HD + d) * S_LEN + s] = vv;
}

// ---------------------------------------------------------------- flash attention
// grid (S/128, NH), 8 waves x 16 query rows; 64-key tiles DMA'd through double-buffered
// LDS by the TDM; online softmax in base-2 domain.
__global__ __launch_bounds__(256)
void k_flash_attn(const unsigned short* __restrict__ Qp,
                  const unsigned short* __restrict__ Kp,
                  const unsigned short* __restrict__ Vt,
                  unsigned short* __restrict__ AOut)       // [S][E] bf16
{
    constexpr int BT   = 64;         // keys per iteration (two WMMA K steps)
    constexpr int LDKC = 40;         // K tile: [row][3 chunks][40] (pad 4DW/16DW)
    constexpr int LDKR = 3 * LDKC;   // 120 elems = 240 B = 60-bank row stride
    constexpr int LDV  = BT + 8;     // 72 elems = 144 B = 36-bank row stride (pad 4DW/32DW)
    constexpr int LDP  = BT + 8;

    __shared__ alignas(16) unsigned short Ks[2][BT * LDKR];    // 64 x 96 (keys x d)
    __shared__ alignas(16) unsigned short Vs[2][HD * LDV];     // 80 x 64 (d x keys)
    __shared__ alignas(16) unsigned short Ps[8 * 16 * LDP];    // per-wave 16 x 64 P tile

    const int tid   = threadIdx.x;
    const int lane  = tid & 31;
    const int wave  = tid >> 5;
    const int laneM = lane & 15;
    const int laneH = lane >> 4;
    const int h     = blockIdx.y;
    const int q0    = blockIdx.x * 128 + wave * 16;

    const unsigned short* Kh = Kp + (size_t)h * S_LEN * HDP;
    const unsigned short* Vh = Vt + (size_t)h * HD * S_LEN;

    // Q fragments for K = 0..95 (3 x v16bf), kept in registers
    FragBF qf[3];
    {
        const unsigned short* qrow = Qp + ((size_t)h * S_LEN + q0 + laneM) * HDP;
#pragma unroll
        for (int kk = 0; kk < 3; ++kk) {
            qf[kk].q[0] = *(const u32x4*)(qrow + kk * 32 + laneH * 8);
            qf[kk].q[1] = *(const u32x4*)(qrow + kk * 32 + laneH * 8 + 16);
        }
    }

    v8f   o[5] = {};
    float mrow[8], lrow[8];
#pragma unroll
    for (int r = 0; r < 8; ++r) { mrow[r] = -1e30f; lrow[r] = 0.0f; }

#if USE_TDM
    if (wave == 0) {    // prologue: DMA key/value tile 0 into buffer 0
        // K tile as 3D: x=32 elems, y=3 chunks (stride 32), z=64 keys (stride HDP)
        tdm_load_tile(Kh, lds_off(&Ks[0][0]), 32, 3, BT, 32, HDP, 3, 3);
        // V tile as 2D: x=64 keys, y=80 d-rows (stride S_LEN)
        tdm_load_tile(Vh, lds_off(&Vs[0][0]), BT, HD, 0, S_LEN, 0, 4, 3);
        __builtin_amdgcn_s_wait_tensorcnt(0);
    }
    __syncthreads();
#endif

    for (int t0 = 0; t0 < S_LEN; t0 += BT) {
        const int buf = (t0 >> 6) & 1;
#if USE_TDM
        if ((t0 + BT < S_LEN) && wave == 0) {    // DMA next tile while computing
            tdm_load_tile(Kh + (size_t)(t0 + BT) * HDP, lds_off(&Ks[buf ^ 1][0]),
                          32, 3, BT, 32, HDP, 3, 3);
            tdm_load_tile(Vh + (t0 + BT), lds_off(&Vs[buf ^ 1][0]),
                          BT, HD, 0, S_LEN, 0, 4, 3);
        }
#else
        __syncthreads();
        {   // fallback staging: K = 768 b128 chunks, V = 640 b128 chunks
            for (int c = tid; c < 768; c += 256) {
                int row = c / 12, b = c % 12, kk = b >> 2, i8 = b & 3;
                *(u32x4*)(&Ks[buf][row * LDKR + kk * LDKC + i8 * 8]) =
                    *(const u32x4*)(Kh + (size_t)(t0 + row) * HDP + b * 8);
            }
            for (int c = tid; c < 640; c += 256) {
                int row = c >> 3, b = c & 7;
                *(u32x4*)(&Vs[buf][row * LDV + b * 8]) =
                    *(const u32x4*)(Vh + (size_t)row * S_LEN + t0 + b * 8);
            }
        }
        __syncthreads();
#endif

        // scores: 16 (q rows) x 64 (keys), four 16x16 C tiles, K = 96
        v8f sc[4] = {};
#pragma unroll
        for (int ct = 0; ct < 4; ++ct) {
#pragma unroll
            for (int kk = 0; kk < 3; ++kk) {
                FragBF bk;
                const unsigned short* p =
                    &Ks[buf][(ct * 16 + laneM) * LDKR + kk * LDKC + laneH * 8];
                bk.q[0] = *(const u32x4*)(p);
                bk.q[1] = *(const u32x4*)(p + 16);
                sc[ct] = __builtin_amdgcn_wmma_f32_16x16x32_bf16(
                    false, qf[kk].v, false, bk.v, (short)0, sc[ct], false, false);
            }
        }

        // online softmax (base-2 domain; scale folded into Q)
        float alpha[8];
#pragma unroll
        for (int r = 0; r < 8; ++r) {
            float v = fmaxf(fmaxf(sc[0][r], sc[1][r]), fmaxf(sc[2][r], sc[3][r]));
            v = fmaxf(v, __shfl_xor(v, 1, 32));
            v = fmaxf(v, __shfl_xor(v, 2, 32));
            v = fmaxf(v, __shfl_xor(v, 4, 32));
            v = fmaxf(v, __shfl_xor(v, 8, 32));
            float mnew = fmaxf(mrow[r], v);
            alpha[r] = __builtin_amdgcn_exp2f(mrow[r] - mnew);
            mrow[r] = mnew;
        }

        unsigned short* pw = Ps + wave * 16 * LDP;     // wave-private P region
#pragma unroll
        for (int r = 0; r < 8; ++r) {
            const int m = r + laneH * 8;
            float p0 = __builtin_amdgcn_exp2f(sc[0][r] - mrow[r]);
            float p1 = __builtin_amdgcn_exp2f(sc[1][r] - mrow[r]);
            float p2 = __builtin_amdgcn_exp2f(sc[2][r] - mrow[r]);
            float p3 = __builtin_amdgcn_exp2f(sc[3][r] - mrow[r]);
            pw[m * LDP + laneM]      = f2bf(p0);
            pw[m * LDP + 16 + laneM] = f2bf(p1);
            pw[m * LDP + 32 + laneM] = f2bf(p2);
            pw[m * LDP + 48 + laneM] = f2bf(p3);
            float ps = (p0 + p1) + (p2 + p3);
            ps += __shfl_xor(ps, 1, 32);
            ps += __shfl_xor(ps, 2, 32);
            ps += __shfl_xor(ps, 4, 32);
            ps += __shfl_xor(ps, 8, 32);
            lrow[r] = lrow[r] * alpha[r] + ps;
        }
#pragma unroll
        for (int dt = 0; dt < 5; ++dt)
#pragma unroll
            for (int r = 0; r < 8; ++r)
                o[dt][r] *= alpha[r];

        // P C-layout -> A-fragments via LDS (K = 64 -> 2 frags)
        FragBF pf[2];
#pragma unroll
        for (int kh = 0; kh < 2; ++kh) {
            const unsigned short* p = pw + laneM * LDP + kh * 32 + laneH * 8;
            pf[kh].q[0] = *(const u32x4*)(p);
            pf[kh].q[1] = *(const u32x4*)(p + 16);
        }
        // AV: 5 d-tiles x 2 K-steps
#pragma unroll
        for (int dt = 0; dt < 5; ++dt) {
#pragma unroll
            for (int kh = 0; kh < 2; ++kh) {
                FragBF bv;
                const unsigned short* p = &Vs[buf][(dt * 16 + laneM) * LDV + kh * 32 + laneH * 8];
                bv.q[0] = *(const u32x4*)(p);
                bv.q[1] = *(const u32x4*)(p + 16);
                o[dt] = __builtin_amdgcn_wmma_f32_16x16x32_bf16(
                    false, pf[kh].v, false, bv.v, (short)0, o[dt], false, false);
            }
        }

#if USE_TDM
        if (wave == 0) __builtin_amdgcn_s_wait_tensorcnt(0);
        __syncthreads();
#endif
    }

#pragma unroll
    for (int r = 0; r < 8; ++r) lrow[r] = 1.0f / lrow[r];
#pragma unroll
    for (int dt = 0; dt < 5; ++dt) {
#pragma unroll
        for (int r = 0; r < 8; ++r) {
            const int m   = q0 + r + laneH * 8;
            const int col = h * HD + dt * 16 + laneM;
            AOut[(size_t)m * EMB + col] = f2bf(o[dt][r] * lrow[r]);
        }
    }
}

// ---------------------------------------------------------------- launch
extern "C" void kernel_launch(void* const* d_in, const int* in_sizes, int n_in,
                              void* d_out, int out_size, void* d_ws, size_t ws_size,
                              hipStream_t stream) {
    const float* hidden = (const float*)d_in[0];
    const float* cosb   = (const float*)d_in[1];
    const float* sinb   = (const float*)d_in[2];
    const float* w_qkv  = (const float*)d_in[3];
    const float* b_qkv  = (const float*)d_in[4];
    const float* w_proj = (const float*)d_in[5];
    const float* b_proj = (const float*)d_in[6];
    (void)in_sizes; (void)n_in; (void)out_size; (void)ws_size;

    char* ws = (char*)d_ws;
    size_t off = 0;
    auto alloc = [&](size_t bytes) {
        char* p = ws + off;
        off += (bytes + 255) & ~(size_t)255;
        return p;
    };
    unsigned short* Xbf   = (unsigned short*)alloc((size_t)S_LEN * EMB * 2);
    unsigned short* Wqkvb = (unsigned short*)alloc((size_t)3 * EMB * EMB * 2);
    unsigned short* Wpb   = (unsigned short*)alloc((size_t)EMB * EMB * 2);
    unsigned short* QKVb  = (unsigned short*)alloc((size_t)S_LEN * 3 * EMB * 2);
    unsigned short* Qp    = (unsigned short*)alloc((size_t)NH * S_LEN * HDP * 2);
    unsigned short* Kpb   = (unsigned short*)alloc((size_t)NH * S_LEN * HDP * 2);
    unsigned short* Vt    = (unsigned short*)alloc((size_t)NH * HD * S_LEN * 2);
    unsigned short* AOut  = (unsigned short*)alloc((size_t)S_LEN * EMB * 2);

    { int n = S_LEN * EMB;     k_f32_to_bf16<<<(n + 255) / 256, 256, 0, stream>>>(hidden, Xbf,   n); }
    { int n = 3 * EMB * EMB;   k_f32_to_bf16<<<(n + 255) / 256, 256, 0, stream>>>(w_qkv,  Wqkvb, n); }
    { int n = EMB * EMB;       k_f32_to_bf16<<<(n + 255) / 256, 256, 0, stream>>>(w_proj, Wpb,   n); }

    // QKV: M=4096, N=3840, K=1280, bf16 out + bias
    k_gemm_bf16<0><<<dim3(3 * EMB / 128, S_LEN / 128), 256, 0, stream>>>(
        Xbf, Wqkvb, b_qkv, nullptr, QKVb, nullptr, S_LEN, 3 * EMB, EMB);

    { int n = S_LEN * EMB;
      k_rope_prep<<<(n + 255) / 256, 256, 0, stream>>>(QKVb, cosb, sinb, Qp, Kpb, Vt); }

    k_flash_attn<<<dim3(S_LEN / 128, NH), 256, 0, stream>>>(Qp, Kpb, Vt, AOut);

    // out-proj: M=4096, N=1280, K=1280, fp32 out + bias + residual
    k_gemm_bf16<1><<<dim3(EMB / 128, S_LEN / 128), 256, 0, stream>>>(
        AOut, Wpb, b_proj, hidden, nullptr, (float*)d_out, S_LEN, EMB, EMB);
}
